// BailingMoEBlock_61263413510573
// MI455X (gfx1250) — matrix operand order repaired and checked by reading, not compiled, not run on hardware
//
#include <hip/hip_runtime.h>
#include <math.h>
#include <stdint.h>

// ---------------- model constants ----------------
constexpr int kH   = 2048;
constexpr int kNH  = 16;
constexpr int kNKV = 4;
constexpr int kDH  = 128;
constexpr int kE   = 8;
constexpr int kI   = 1024;
constexpr int kIS  = 1024;
constexpr int kS   = 1024;   // B*S tokens
constexpr float kEps = 1e-6f;

typedef __attribute__((ext_vector_type(16))) _Float16 v16h;
typedef __attribute__((ext_vector_type(8)))  float    v8f;

union HV2 { uint4 u[2]; v16h v; };

// ---------------- WMMA helper (CDNA5 16x16x32 f16) ----------------
__device__ __forceinline__ v8f wmma_f16(v16h a, v16h b, v8f c) {
  // (neg_a, A, neg_b, B, c_mod, C, reuse_a, reuse_b)
  return __builtin_amdgcn_wmma_f32_16x16x32_f16(false, a, false, b, (short)0, c, false, false);
}

// A-frag (16x32 f16, ISA 7.12.2): lane L row M=L%16; elems 0-7 K=(L/16)*8+i, elems 8-15 K=16+(L/16)*8+i
// Source: row-major [M][K] LDS tile. Two aligned 16B chunks.
__device__ __forceinline__ v16h load_a_frag(const _Float16* s, int pitch, int m0, int lane) {
  HV2 h;
  const _Float16* p = s + (m0 + (lane & 15)) * pitch + ((lane >> 4) << 3);
  h.u[0] = *(const uint4*)p;
  h.u[1] = *(const uint4*)(p + 16);
  return h.v;
}

// B-frag (32x16 f16): lane L col N=L%16; elems 0-7 K=(L/16)*8+i, elems 8-15 K=16+(L/16)*8+i
// Source: col-major [N][K] LDS tile -> two aligned 16B chunks.
__device__ __forceinline__ v16h load_bT_frag(const _Float16* s, int pitch, int n0, int lane) {
  HV2 h;
  const _Float16* p = s + (n0 + (lane & 15)) * pitch + ((lane >> 4) << 3);
  h.u[0] = *(const uint4*)p;
  h.u[1] = *(const uint4*)(p + 16);
  return h.v;
}

// ---------------- RMSNorm -> f16 ----------------
__global__ __launch_bounds__(256)
void rmsnorm_to_f16(const float* __restrict__ x, const float* __restrict__ w,
                    _Float16* __restrict__ out, int ncols) {
  int row = blockIdx.x;
  const float* xr = x + (size_t)row * ncols;
  float ss = 0.f;
  for (int c = threadIdx.x; c < ncols; c += blockDim.x) { float v = xr[c]; ss += v * v; }
#pragma unroll
  for (int m = 16; m >= 1; m >>= 1) ss += __shfl_xor(ss, m, 32);
  __shared__ float parts[8];
  int wid = threadIdx.x >> 5, lane = threadIdx.x & 31;
  if (lane == 0) parts[wid] = ss;
  __syncthreads();
  float tot = 0.f;
#pragma unroll
  for (int i = 0; i < 8; ++i) tot += parts[i];
  float rs = rsqrtf(tot / (float)ncols + kEps);
  _Float16* o = out + (size_t)row * ncols;
  for (int c = threadIdx.x; c < ncols; c += blockDim.x)
    o[c] = (_Float16)(xr[c] * rs * w[c]);
}

// ---------------- Generic WMMA GEMM: C[M,N] = A16[M,K] * B32[K,N] (+ addend) ----------------
constexpr int BM = 128, BN = 64, BK = 32;
constexpr int APITCH = 40;   // halves: 80B rows, 16B aligned, bank-stride 20 (conflict-free)
constexpr int BKP    = 40;   // halves: col-major B tile [N][K]

__global__ __launch_bounds__(256)
void gemm_f16f32(const _Float16* __restrict__ A, const float* __restrict__ B,
                 const float* __restrict__ addend, float* __restrict__ C,
                 int M, int N, int K) {
  __shared__ _Float16 As[BM * APITCH];
  __shared__ _Float16 Bs[BN * BKP];      // col-major [n][k]
  int tid = threadIdx.x;
  int lane = tid & 31, wave = tid >> 5;
  int wm = wave >> 1, wn = wave & 1;     // 4x2 wave grid, 32x32 per wave
  int bm = blockIdx.y * BM, bn = blockIdx.x * BN;
  v8f acc[2][2] = {};

  int arow = tid >> 1, achunk = (tid & 1) * 16;   // A: 128 rows x 2 chunks of 16 halves
  int bcol = tid & 63, bkg = (tid >> 6) << 3;     // B: per-thread 8-deep K column of one N

  unsigned lds_a = (unsigned)(uintptr_t)(&As[arow * APITCH + achunk]);

  for (int kk = 0; kk < K; kk += BK) {
    // A tile: HBM -> LDS via CDNA5 async DMA (two b128 per thread, ASYNCcnt tracked)
    unsigned long long ag =
        (unsigned long long)(uintptr_t)(A + (size_t)(bm + arow) * K + kk + achunk);
    asm volatile("global_load_async_to_lds_b128 %0, %1, off"
                 :: "v"(lds_a), "v"(ag) : "memory");
    asm volatile("global_load_async_to_lds_b128 %0, %1, off"
                 :: "v"(lds_a + 16u), "v"(ag + 16ull) : "memory");

    // B tile: coalesced column reads (lane<->N), cvt f32->f16, one b128 store (col-major)
    const float* bg = B + (size_t)(kk + bkg) * N + bn + bcol;
    float bv[8];
#pragma unroll
    for (int j = 0; j < 8; ++j) bv[j] = bg[(size_t)j * N];
    if (kk + BK < K) __builtin_prefetch(bg + (size_t)BK * N, 0, 1);
    alignas(16) _Float16 h8[8];
#pragma unroll
    for (int j = 0; j < 8; ++j) h8[j] = (_Float16)bv[j];
    *(uint4*)&Bs[bcol * BKP + bkg] = *(const uint4*)h8;

    asm volatile("s_wait_asynccnt 0" ::: "memory");
    __syncthreads();

    v16h af[2], bf[2];
    af[0] = load_a_frag(As, APITCH, wm * 32,      lane);
    af[1] = load_a_frag(As, APITCH, wm * 32 + 16, lane);
    bf[0] = load_bT_frag(Bs, BKP,   wn * 32,      lane);
    bf[1] = load_bT_frag(Bs, BKP,   wn * 32 + 16, lane);
#pragma unroll
    for (int mi = 0; mi < 2; ++mi)
#pragma unroll
      for (int ni = 0; ni < 2; ++ni)
        acc[mi][ni] = wmma_f16(af[mi], bf[ni], acc[mi][ni]);
    __syncthreads();
  }

  // epilogue: C/D layout — VGPR r: lanes 0-15 -> M=r, lanes 16-31 -> M=r+8; N=lane%16
#pragma unroll
  for (int mi = 0; mi < 2; ++mi)
#pragma unroll
    for (int ni = 0; ni < 2; ++ni) {
      int rbase = bm + wm * 32 + mi * 16 + ((lane >> 4) << 3);
      int col   = bn + wn * 32 + ni * 16 + (lane & 15);
#pragma unroll
      for (int r = 0; r < 8; ++r) {
        size_t idx = (size_t)(rbase + r) * N + col;
        float v = acc[mi][ni][r];
        if (addend) v += addend[idx];
        C[idx] = v;
      }
    }
}

// ---------------- per-head RMSNorm + NeoX RoPE + f16 pack ----------------
// q16: [NH][S][DH], k16: [NKV][S][DH], v16t: [NKV][DH][S] (transposed for WMMA B-frags)
__global__ __launch_bounds__(128)
void qkv_norm_rope(const float* __restrict__ qkv, const float* __restrict__ qw,
                   const float* __restrict__ kw, const int* __restrict__ pos,
                   _Float16* __restrict__ q16, _Float16* __restrict__ k16,
                   _Float16* __restrict__ v16t) {
  int s = blockIdx.x, hh = blockIdx.y, d = threadIdx.x;
  __shared__ float row[kDH];
  __shared__ float parts[4];
  const float* src = qkv + (size_t)s * ((kNH + 2 * kNKV) * kDH) + hh * kDH;
  float v = src[d];
  if (hh >= kNH + kNKV) {                       // V head: f16 convert, transposed store
    v16t[((size_t)(hh - kNH - kNKV) * kDH + d) * kS + s] = (_Float16)v;
    return;
  }
  float ss = v * v;
#pragma unroll
  for (int m = 16; m >= 1; m >>= 1) ss += __shfl_xor(ss, m, 32);
  if ((d & 31) == 0) parts[d >> 5] = ss;
  __syncthreads();
  float tot = parts[0] + parts[1] + parts[2] + parts[3];
  float rs = rsqrtf(tot / (float)kDH + kEps);
  const float* w = (hh < kNH) ? qw : kw;
  row[d] = v * rs * w[d];
  __syncthreads();
  int i = d & 63;
  float freq = (float)pos[s] * __powf(10000.f, -2.f * (float)i / (float)kDH);
  float c = __cosf(freq), sn = __sinf(freq);
  float out = (d < 64) ? (row[d] * c - row[d + 64] * sn)
                       : (row[d] * c + row[d - 64] * sn);
  if (hh < kNH) {
    out *= 0.08838834764831845f;               // DH^-0.5 folded into Q
    q16[((size_t)hh * kS + s) * kDH + d] = (_Float16)out;
  } else {
    k16[((size_t)(hh - kNH) * kS + s) * kDH + d] = (_Float16)out;
  }
}

// ---------------- flash-style causal GQA attention (WMMA) ----------------
constexpr int KP = 136;  // K-tile pitch (halves): 272B rows -> bank stride 4, conflict-free
constexpr int VP = 40;   // V-tile pitch (halves): [d][key] layout

__global__ __launch_bounds__(128)
void attn_kernel(const _Float16* __restrict__ q16,  // [NH][S][DH]
                 const _Float16* __restrict__ k16,  // [NKV][S][DH]
                 const _Float16* __restrict__ v16t, // [NKV][DH][S]
                 _Float16* __restrict__ ctx16) {    // [S][NH*DH]
  __shared__ _Float16 Ks[32 * KP];    // [key][d]
  __shared__ _Float16 Vs[kDH * VP];   // [d][key]
  __shared__ _Float16 Ps[4][16 * 32];
  int h = blockIdx.y, kvh = h >> 2;
  int qb0 = blockIdx.x * 64;
  int tid = threadIdx.x, lane = tid & 31, wave = tid >> 5;
  int q0 = qb0 + wave * 16;
  int kbse = (lane >> 4) << 3;

  // Q fragments: 4 K-chunks of 32 over DH=128 (aligned b128 pairs from global)
  v16h qf[4];
  const _Float16* qb = q16 + ((size_t)h * kS + q0 + (lane & 15)) * kDH + kbse;
#pragma unroll
  for (int c = 0; c < 4; ++c) {
    HV2 hq;
    hq.u[0] = *(const uint4*)(qb + c * 32);
    hq.u[1] = *(const uint4*)(qb + c * 32 + 16);
    qf[c] = hq.v;
  }

  v8f o[8] = {};
  float mrun[8], lrun[8];
#pragma unroll
  for (int r = 0; r < 8; ++r) { mrun[r] = -1e30f; lrun[r] = 0.f; }

  int nkb = (qb0 + 64) >> 5;
  for (int kb = 0; kb < nkb; ++kb) {
    int krow0 = kb * 32;
    {  // K tile: [key][d] row-major, 4x b128 per thread
      int row = tid >> 2, chunk = (tid & 3) * 32;
      const uint4* ks = (const uint4*)(k16 + ((size_t)kvh * kS + krow0 + row) * kDH + chunk);
      uint4* kd = (uint4*)&Ks[row * KP + chunk];
#pragma unroll
      for (int i = 0; i < 4; ++i) kd[i] = ks[i];
    }
    {  // V tile: [d][key] straight from transposed global, 4x b128 per thread
      int d = tid;
      const uint4* vs = (const uint4*)(v16t + ((size_t)kvh * kDH + d) * kS + krow0);
      uint4* vd = (uint4*)&Vs[d * VP];
#pragma unroll
      for (int i = 0; i < 4; ++i) vd[i] = vs[i];
    }
    __syncthreads();

    if (krow0 <= q0 + 15) {
      // scores: S = Q Kt (two 16-key tiles); B-frag = contiguous d-run of Ks
      v8f sc[2] = {};
#pragma unroll
      for (int c = 0; c < 4; ++c) {
        int kd0 = c * 32 + ((lane >> 4) << 3);
        HV2 b0, b1;
        const _Float16* kp0 = &Ks[(lane & 15) * KP + kd0];
        const _Float16* kp1 = &Ks[(16 + (lane & 15)) * KP + kd0];
        b0.u[0] = *(const uint4*)kp0; b0.u[1] = *(const uint4*)(kp0 + 16);
        b1.u[0] = *(const uint4*)kp1; b1.u[1] = *(const uint4*)(kp1 + 16);
        sc[0] = wmma_f16(qf[c], b0.v, sc[0]);
        sc[1] = wmma_f16(qf[c], b1.v, sc[1]);
      }
      // online softmax over C-layout rows
      int qoff = (lane >> 4) << 3;
      int keyc = lane & 15;
#pragma unroll
      for (int r = 0; r < 8; ++r) {
        int q  = q0 + r + qoff;
        int k0 = krow0 + keyc, k1 = krow0 + 16 + keyc;
        float s0 = (k0 <= q) ? sc[0][r] : -1e30f;
        float s1 = (k1 <= q) ? sc[1][r] : -1e30f;
        float rmax = fmaxf(s0, s1);
#pragma unroll
        for (int m = 8; m >= 1; m >>= 1) rmax = fmaxf(rmax, __shfl_xor(rmax, m, 32));
        float mnew = fmaxf(mrun[r], rmax);
        float p0 = (k0 <= q) ? __expf(s0 - mnew) : 0.f;
        float p1 = (k1 <= q) ? __expf(s1 - mnew) : 0.f;
        float rsum = p0 + p1;
#pragma unroll
        for (int m = 8; m >= 1; m >>= 1) rsum += __shfl_xor(rsum, m, 32);
        float alpha = __expf(mrun[r] - mnew);
        lrun[r] = lrun[r] * alpha + rsum;
        mrun[r] = mnew;
#pragma unroll
        for (int j = 0; j < 8; ++j) o[j][r] *= alpha;
        int prow = r + qoff;
        Ps[wave][prow * 32 + keyc]      = (_Float16)p0;
        Ps[wave][prow * 32 + 16 + keyc] = (_Float16)p1;
      }
      // P as A-frag (16x32); O += P @ V  (Vs is [d][key] -> contiguous key runs)
      HV2 hp;
      const _Float16* pp = &Ps[wave][(lane & 15) * 32 + kbse];
      hp.u[0] = *(const uint4*)pp;
      hp.u[1] = *(const uint4*)(pp + 16);
#pragma unroll
      for (int j = 0; j < 8; ++j) {
        HV2 bv;
        const _Float16* vp = &Vs[(j * 16 + (lane & 15)) * VP + ((lane >> 4) << 3)];
        bv.u[0] = *(const uint4*)vp;
        bv.u[1] = *(const uint4*)(vp + 16);
        o[j] = wmma_f16(hp.v, bv.v, o[j]);
      }
    }
    __syncthreads();
  }

  int qoff = (lane >> 4) << 3;
#pragma unroll
  for (int r = 0; r < 8; ++r) {
    int token = q0 + r + qoff;
    float inv = 1.f / lrun[r];
#pragma unroll
    for (int j = 0; j < 8; ++j)
      ctx16[(size_t)token * (kNH * kDH) + h * kDH + j * 16 + (lane & 15)] =
          (_Float16)(o[j][r] * inv);
  }
}

// ---------------- router: softmax -> top-2 -> renorm combine weights ----------------
__global__ __launch_bounds__(256)
void router_kernel(const _Float16* __restrict__ h2, const float* __restrict__ wr,
                   float* __restrict__ comb) {
  int t = blockIdx.x;
  int wave = threadIdx.x >> 5, lane = threadIdx.x & 31;   // wave == expert
  __shared__ float logits[kE];
  const _Float16* hr = h2 + (size_t)t * kH;
  const float* wre = wr + (size_t)wave * kH;
  float sum = 0.f;
  for (int c = lane; c < kH; c += 32) sum += (float)hr[c] * wre[c];
#pragma unroll
  for (int m = 16; m >= 1; m >>= 1) sum += __shfl_xor(sum, m, 32);
  if (lane == 0) logits[wave] = sum;
  __syncthreads();
  if (threadIdx.x == 0) {
    float mx = logits[0];
    for (int e = 1; e < kE; ++e) mx = fmaxf(mx, logits[e]);
    float p[kE], Z = 0.f;
    for (int e = 0; e < kE; ++e) { p[e] = __expf(logits[e] - mx); Z += p[e]; }
    for (int e = 0; e < kE; ++e) p[e] /= Z;
    int i0 = 0;
    for (int e = 1; e < kE; ++e) if (p[e] > p[i0]) i0 = e;
    int i1 = (i0 == 0) ? 1 : 0;
    for (int e = 0; e < kE; ++e) if (e != i0 && p[e] > p[i1]) i1 = e;
    float wsum = p[i0] + p[i1];
    float* cr = comb + (size_t)t * kE;
    for (int e = 0; e < kE; ++e) cr[e] = 0.f;
    cr[i0] = p[i0] / wsum;
    cr[i1] = p[i1] / wsum;
  }
}

// ---------------- silu(g)*u * comb -> f16 ----------------
__global__ __launch_bounds__(256)
void silu_mul_scale(const float* __restrict__ gu, const float* __restrict__ comb,
                    int expert, _Float16* __restrict__ act) {
  int idx = blockIdx.x * blockDim.x + threadIdx.x;
  int t = idx >> 10, i = idx & (kI - 1);
  float g = gu[(size_t)t * (2 * kI) + i];
  float u = gu[(size_t)t * (2 * kI) + kI + i];
  float sg = g / (1.f + __expf(-g));
  float scale = comb ? comb[(size_t)t * kE + expert] : 1.f;
  act[idx] = (_Float16)(sg * u * scale);
}

// ---------------- out = a + b ----------------
__global__ __launch_bounds__(256)
void add2_kernel(const float* __restrict__ a, const float* __restrict__ b,
                 float* __restrict__ out, int n) {
  int i = blockIdx.x * blockDim.x + threadIdx.x;
  if (i < n) out[i] = a[i] + b[i];
}

// ---------------- host orchestration ----------------
extern "C" void kernel_launch(void* const* d_in, const int* in_sizes, int n_in,
                              void* d_out, int out_size, void* d_ws, size_t ws_size,
                              hipStream_t stream) {
  (void)in_sizes; (void)n_in; (void)out_size; (void)ws_size;
  const float* hidden   = (const float*)d_in[0];
  const float* w_ln1    = (const float*)d_in[1];
  const float* w_ln2    = (const float*)d_in[2];
  const float* wqkv     = (const float*)d_in[3];
  const float* q_norm_w = (const float*)d_in[4];
  const float* k_norm_w = (const float*)d_in[5];
  const float* wo       = (const float*)d_in[6];
  const float* w_router = (const float*)d_in[7];
  const float* w_gu     = (const float*)d_in[8];
  const float* w_d      = (const float*)d_in[9];
  const float* ws_gu    = (const float*)d_in[10];
  const float* ws_d     = (const float*)d_in[11];
  const int*   positions= (const int*)d_in[12];
  float* out = (float*)d_out;

  char* wsp = (char*)d_ws;
  size_t off = 0;
  auto take = [&](size_t bytes) -> void* {
    void* p = wsp + off;
    off += (bytes + 255) & ~(size_t)255;
    return p;
  };
  const int kQKVN = (kNH + 2 * kNKV) * kDH;  // 3072
  _Float16* h16    = (_Float16*)take((size_t)kS * kH * 2);
  float*    qkv    = (float*)   take((size_t)kS * kQKVN * 4);
  _Float16* q16    = (_Float16*)take((size_t)kNH * kS * kDH * 2);
  _Float16* k16    = (_Float16*)take((size_t)kNKV * kS * kDH * 2);
  _Float16* v16t   = (_Float16*)take((size_t)kNKV * kDH * kS * 2);
  _Float16* ctx16  = (_Float16*)take((size_t)kS * kH * 2);
  float*    xres   = (float*)   take((size_t)kS * kH * 4);
  _Float16* h2     = (_Float16*)take((size_t)kS * kH * 2);
  float*    comb   = (float*)   take((size_t)kS * kE * 4);
  float*    gu     = (float*)   take((size_t)kS * 2 * kI * 4);
  _Float16* act16  = (_Float16*)take((size_t)kS * kI * 2);
  float*    routed = (float*)   take((size_t)kS * kH * 4);

  // ---- attention sub-block ----
  rmsnorm_to_f16<<<kS, 256, 0, stream>>>(hidden, w_ln1, h16, kH);
  gemm_f16f32<<<dim3(kQKVN / BN, kS / BM), 256, 0, stream>>>(h16, wqkv, nullptr, qkv, kS, kQKVN, kH);
  qkv_norm_rope<<<dim3(kS, kNH + 2 * kNKV), 128, 0, stream>>>(qkv, q_norm_w, k_norm_w, positions,
                                                              q16, k16, v16t);
  attn_kernel<<<dim3(kS / 64, kNH), 128, 0, stream>>>(q16, k16, v16t, ctx16);
  gemm_f16f32<<<dim3(kH / BN, kS / BM), 256, 0, stream>>>(ctx16, wo, hidden, xres, kS, kH, kH);

  // ---- MoE sub-block ----
  rmsnorm_to_f16<<<kS, 256, 0, stream>>>(xres, w_ln2, h2, kH);
  router_kernel<<<kS, 256, 0, stream>>>(h2, w_router, comb);
  for (int e = 0; e < kE; ++e) {
    const float* wgu_e = w_gu + (size_t)e * kH * 2 * kI;
    const float* wd_e  = w_d  + (size_t)e * kI * kH;
    gemm_f16f32<<<dim3(2 * kI / BN, kS / BM), 256, 0, stream>>>(h2, wgu_e, nullptr, gu, kS, 2 * kI, kH);
    silu_mul_scale<<<(kS * kI) / 256, 256, 0, stream>>>(gu, comb, e, act16);
    gemm_f16f32<<<dim3(kH / BN, kS / BM), 256, 0, stream>>>(act16, wd_e,
                                                            (e == 0) ? nullptr : routed, routed,
                                                            kS, kH, kI);
  }
  // shared expert (weight 1), accumulated into routed
  gemm_f16f32<<<dim3(2 * kIS / BN, kS / BM), 256, 0, stream>>>(h2, ws_gu, nullptr, gu, kS, 2 * kIS, kH);
  silu_mul_scale<<<(kS * kIS) / 256, 256, 0, stream>>>(gu, nullptr, 0, act16);
  gemm_f16f32<<<dim3(kH / BN, kS / BM), 256, 0, stream>>>(act16, ws_d, routed, routed, kS, kH, kIS);

  // final residual
  add2_kernel<<<(kS * kH) / 256, 256, 0, stream>>>(xres, routed, out, kS * kH);
}
